// Attention_2662879723851
// MI455X (gfx1250) — matrix operand order repaired and checked by reading, not compiled, not run on hardware
//
#include <hip/hip_runtime.h>
#include <hip/hip_bf16.h>

// ---------------------------------------------------------------------------
// Types for CDNA5 WMMA (wave32):  D(16x16 f32) = A(16x32 bf16) * B(32x16 bf16) + C
// ---------------------------------------------------------------------------
typedef __attribute__((ext_vector_type(16))) __bf16 v16bf;
typedef __attribute__((ext_vector_type(8)))  __bf16 v8bf;
typedef __attribute__((ext_vector_type(8)))  float  v8f;

#define SEQ_N   2048
#define DIMH    64
#define NHEADS  16
#define WIN     256

static __device__ __forceinline__ v8bf ld8(const __bf16* p) {
    return *(const v8bf*)p;
}
static __device__ __forceinline__ v16bf cat8(v8bf lo, v8bf hi) {
    return __builtin_shufflevector(lo, hi, 0,1,2,3,4,5,6,7,8,9,10,11,12,13,14,15);
}
static __device__ __forceinline__ v8f vzero8() {
    v8f v = {0.f,0.f,0.f,0.f,0.f,0.f,0.f,0.f};
    return v;
}
static __device__ __forceinline__ v8f wmma_bf16(v16bf a, v16bf b, v8f c) {
    // 8 args: (neg_a, A, neg_b, B, c_mod, C, reuse_a, reuse_b)
    return __builtin_amdgcn_wmma_f32_16x16x32_bf16(false, a, false, b, (short)0, c,
                                                   false, false);
}

// ---------------------------------------------------------------------------
// Stage 0a: fp32 -> bf16 cast (same layout)
// ---------------------------------------------------------------------------
__global__ void cast_bf16_kernel(const float* __restrict__ in,
                                 __bf16* __restrict__ out, int n) {
    int i = blockIdx.x * blockDim.x + threadIdx.x;
    int stride = gridDim.x * blockDim.x;
    for (; i < n; i += stride) out[i] = (__bf16)in[i];
}

// ---------------------------------------------------------------------------
// Stage 0b: fp32 [R][C] -> bf16 transposed [C][R]
// ---------------------------------------------------------------------------
__global__ void transpose_cast_kernel(const float* __restrict__ in,
                                      __bf16* __restrict__ out, int R, int C) {
    int idx = blockIdx.x * blockDim.x + threadIdx.x;
    if (idx >= R * C) return;
    int r = idx % R;          // fastest -> contiguous bf16 writes
    int c = idx / R;
    out[(size_t)c * R + r] = (__bf16)in[(size_t)r * C + c];
}

// ---------------------------------------------------------------------------
// Stage 1/5: bf16 GEMM  C[M,N] = A[M,K] * Bt[N,K]^T   (Bt is B transposed)
// 256 threads = 8 waves; each wave computes a 32x32 tile (4 WMMA tiles).
// Block covers 128(M) x 64(N).  M%128==0, N%64==0, K%32==0 assumed.
// Writes f32 (Cf) or bf16 (Cb) output.
// ---------------------------------------------------------------------------
__global__ void __launch_bounds__(256)
gemm_bf16_kernel(const __bf16* __restrict__ A, const __bf16* __restrict__ Bt,
                 float* __restrict__ Cf, __bf16* __restrict__ Cb,
                 int M, int N, int K) {
    const int lane = threadIdx.x & 31;
    const int w    = threadIdx.x >> 5;
    const int wm   = w & 3;
    const int wn   = w >> 2;
    const int m0   = blockIdx.x * 128 + wm * 32;
    const int n0   = blockIdx.y * 64  + wn * 32;
    const int ln   = lane & 15;
    const int half = lane >> 4;

    // A layout (16-bit, 16x32): lane ln holds row M=ln; half0 slots: K 0..7,16..23
    // half1 slots: K 8..15,24..31.
    const __bf16* a0p = A + (size_t)(m0 + ln) * K + 8 * half;
    const __bf16* a1p = A + (size_t)(m0 + 16 + ln) * K + 8 * half;
    // B layout (32x16): lane ln holds col N=ln; slots = K 8*half + 0..7, 16 + 8*half + 0..7.
    const __bf16* b0p = Bt + (size_t)(n0 + ln) * K + 8 * half;
    const __bf16* b1p = Bt + (size_t)(n0 + 16 + ln) * K + 8 * half;

    v8f c00 = vzero8(), c01 = vzero8(), c10 = vzero8(), c11 = vzero8();

    for (int k0 = 0; k0 < K; k0 += 32) {
        __builtin_prefetch(a0p + k0 + 128, 0, 3);   // global_prefetch_b8
        v16bf a0 = cat8(ld8(a0p + k0), ld8(a0p + k0 + 16));
        v16bf a1 = cat8(ld8(a1p + k0), ld8(a1p + k0 + 16));
        v16bf b0 = cat8(ld8(b0p + k0), ld8(b0p + k0 + 16));
        v16bf b1 = cat8(ld8(b1p + k0), ld8(b1p + k0 + 16));
        c00 = wmma_bf16(a0, b0, c00);
        c01 = wmma_bf16(a0, b1, c01);
        c10 = wmma_bf16(a1, b0, c10);
        c11 = wmma_bf16(a1, b1, c11);
    }

    // C layout: element (vgpr r, lane) -> row = r + 8*half, col = ln.
    #pragma unroll
    for (int t = 0; t < 4; ++t) {
        const v8f c = (t == 0) ? c00 : (t == 1) ? c01 : (t == 2) ? c10 : c11;
        const int mt = m0 + ((t >> 1) * 16);
        const int nt = n0 + ((t & 1) * 16);
        #pragma unroll
        for (int r = 0; r < 8; ++r) {
            const size_t idx = (size_t)(mt + r + 8 * half) * N + nt + ln;
            if (Cf) Cf[idx] = c[r];
            else    Cb[idx] = (__bf16)c[r];
        }
    }
}

// ---------------------------------------------------------------------------
// Stage 2: RoPE + head scatter.
// qkvb: bf16 [4096][3072]; rot: f32 [2048][64]
// Qb,Kb: bf16 [B*H][N][64] (Q pre-scaled by 1/sqrt(64)); Vt: bf16 [B*H][64][N]
// ---------------------------------------------------------------------------
__global__ void rope_scatter_kernel(const __bf16* __restrict__ qkvb,
                                    const float* __restrict__ rot,
                                    __bf16* __restrict__ Qb,
                                    __bf16* __restrict__ Kb,
                                    __bf16* __restrict__ Vt) {
    const int d   = threadIdx.x;          // 0..63
    const int h   = blockIdx.y;           // 0..15
    const int row = blockIdx.x;           // 0..4095
    const int b   = row >> 11;
    const int n   = row & (SEQ_N - 1);

    const __bf16* base = qkvb + (size_t)row * 3072 + h * DIMH;
    const float cs = cosf(rot[n * DIMH + d]);
    const float sn = sinf(rot[n * DIMH + d]);
    const int   pd  = (d < 32) ? d + 32 : d - 32;
    const float sgn = (d < 32) ? -1.f : 1.f;

    const float q  = (float)base[d];
    const float qp = sgn * (float)base[pd];
    const float k  = (float)base[1024 + d];
    const float kp = sgn * (float)base[1024 + pd];
    const float v  = (float)base[2048 + d];

    const int bh = b * NHEADS + h;
    const size_t o = ((size_t)bh * SEQ_N + n) * DIMH + d;
    Qb[o] = (__bf16)((q * cs + qp * sn) * 0.125f);   // scale = 1/sqrt(64)
    Kb[o] = (__bf16)(k * cs + kp * sn);
    Vt[((size_t)bh * DIMH + d) * SEQ_N + n] = (__bf16)v;
}

// ---------------------------------------------------------------------------
// Stage 3: banded flash attention.
// Block = 128 threads (4 waves); each wave handles 16 queries of a 64-query
// block. Unified key range per block -> identical trip counts (barrier-safe).
// Per 32-key step: 4 WMMAs (scores, Dh split 32+32) + 4 WMMAs (PV, Dh tiles).
// ---------------------------------------------------------------------------
__global__ void __launch_bounds__(128)
attn_kernel(const __bf16* __restrict__ Qb, const __bf16* __restrict__ Kb,
            const __bf16* __restrict__ Vt, __bf16* __restrict__ attnb) {
    __shared__ __align__(16) __bf16 Pl[4][16][32];   // per-wave P staging

    const int lane = threadIdx.x & 31;
    const int w    = threadIdx.x >> 5;
    const int ln   = lane & 15;
    const int half = lane >> 4;
    const int qblk = blockIdx.x * 64;
    const int q0   = qblk + w * 16;
    const int bh   = blockIdx.y;
    const int b    = bh >> 4;
    const int h    = bh & 15;

    const __bf16* Q  = Qb + (size_t)bh * SEQ_N * DIMH;
    const __bf16* Kp = Kb + (size_t)bh * SEQ_N * DIMH;
    const __bf16* Vp = Vt + (size_t)bh * DIMH * SEQ_N;

    // Q A-fragments: d 0..31 and d 32..63.
    const __bf16* qrow = Q + (size_t)(q0 + ln) * DIMH + 8 * half;
    const v16bf qa0 = cat8(ld8(qrow),      ld8(qrow + 16));
    const v16bf qa1 = cat8(ld8(qrow + 32), ld8(qrow + 48));

    v8f o0 = vzero8(), o1 = vzero8(), o2 = vzero8(), o3 = vzero8();
    float mrow[8], lrow[8];
    #pragma unroll
    for (int r = 0; r < 8; ++r) { mrow[r] = -1e30f; lrow[r] = 0.f; }
    const int rbase = q0 + 8 * half;

    const int kk_start = (qblk >= WIN) ? (qblk - WIN) : 0;   // 64-aligned
    const int kk_end   = qblk + 64;

    for (int kk = kk_start; kk < kk_end; kk += 32) {
        // K^T B-fragments: score tile 0 = keys kk..kk+15, tile 1 = kk+16..kk+31.
        const __bf16* k0row = Kp + (size_t)(kk + ln) * DIMH + 8 * half;
        const __bf16* k1row = Kp + (size_t)(kk + 16 + ln) * DIMH + 8 * half;
        const v16bf kb00 = cat8(ld8(k0row),      ld8(k0row + 16));  // d 0..31
        const v16bf kb01 = cat8(ld8(k0row + 32), ld8(k0row + 48));  // d 32..63
        const v16bf kb10 = cat8(ld8(k1row),      ld8(k1row + 16));
        const v16bf kb11 = cat8(ld8(k1row + 32), ld8(k1row + 48));

        v8f s0 = vzero8(), s1 = vzero8();
        s0 = wmma_bf16(qa0, kb00, s0);
        s0 = wmma_bf16(qa1, kb01, s0);
        s1 = wmma_bf16(qa0, kb10, s1);
        s1 = wmma_bf16(qa1, kb11, s1);

        // Online softmax in C layout: row = rbase + r, cols = kk+ln / kk+16+ln.
        const int col0 = kk + ln;
        const int col1 = col0 + 16;
        #pragma unroll
        for (int r = 0; r < 8; ++r) {
            const int row = rbase + r;
            const bool v0 = (col0 <= row) && (col0 + WIN >= row);
            const bool v1 = (col1 <= row) && (col1 + WIN >= row);
            float sl = v0 ? s0[r] : -1e30f;
            float sh = v1 ? s1[r] : -1e30f;
            float mx = fmaxf(sl, sh);
            mx = fmaxf(mx, __shfl_xor(mx, 1, 32));
            mx = fmaxf(mx, __shfl_xor(mx, 2, 32));
            mx = fmaxf(mx, __shfl_xor(mx, 4, 32));
            mx = fmaxf(mx, __shfl_xor(mx, 8, 32));
            const float mn = fmaxf(mrow[r], mx);
            const float pl = v0 ? __expf(sl - mn) : 0.f;
            const float ph = v1 ? __expf(sh - mn) : 0.f;
            const float cf = __expf(mrow[r] - mn);
            float sm = pl + ph;
            sm += __shfl_xor(sm, 1, 32);
            sm += __shfl_xor(sm, 2, 32);
            sm += __shfl_xor(sm, 4, 32);
            sm += __shfl_xor(sm, 8, 32);
            lrow[r] = lrow[r] * cf + sm;
            mrow[r] = mn;
            o0[r] *= cf; o1[r] *= cf; o2[r] *= cf; o3[r] *= cf;
            Pl[w][r + 8 * half][ln]      = (__bf16)pl;
            Pl[w][r + 8 * half][16 + ln] = (__bf16)ph;
        }
        __syncthreads();

        // P A-fragment (16x32) from LDS in the documented A layout.
        const __bf16* prow = &Pl[w][ln][8 * half];
        const v16bf pa = cat8(ld8(prow), ld8(prow + 16));

        // V B-fragments from transposed V: keys 8*half+0..7 and 16+8*half+0..7 per lane.
        #pragma unroll
        for (int nt = 0; nt < 4; ++nt) {
            const __bf16* vr = Vp + (size_t)(nt * 16 + ln) * SEQ_N + kk + 8 * half;
            const v16bf vf = cat8(ld8(vr), ld8(vr + 16));
            if      (nt == 0) o0 = wmma_bf16(pa, vf, o0);
            else if (nt == 1) o1 = wmma_bf16(pa, vf, o1);
            else if (nt == 2) o2 = wmma_bf16(pa, vf, o2);
            else              o3 = wmma_bf16(pa, vf, o3);
        }
        __syncthreads();
    }

    // Normalize and store as bf16 rows of [B, N, H*Dh].
    #pragma unroll
    for (int r = 0; r < 8; ++r) {
        const float inv = 1.f / lrow[r];
        const int row = rbase + r;
        const size_t base = ((size_t)b * SEQ_N + row) * 1024 + h * DIMH + ln;
        attnb[base]      = (__bf16)(o0[r] * inv);
        attnb[base + 16] = (__bf16)(o1[r] * inv);
        attnb[base + 32] = (__bf16)(o2[r] * inv);
        attnb[base + 48] = (__bf16)(o3[r] * inv);
    }
}

// ---------------------------------------------------------------------------
// Host-side orchestration
// ---------------------------------------------------------------------------
extern "C" void kernel_launch(void* const* d_in, const int* in_sizes, int n_in,
                              void* d_out, int out_size, void* d_ws, size_t ws_size,
                              hipStream_t stream) {
    const float* x    = (const float*)d_in[0];   // [2,2048,1024]
    const float* rot  = (const float*)d_in[1];   // [2048,64]
    const float* Wqkv = (const float*)d_in[2];   // [1024,3072]
    const float* Wout = (const float*)d_in[3];   // [1024,1024]
    float* out = (float*)d_out;                  // [2,2048,1024]

    const int M = 4096, DIM = 1024, N3 = 3072;

    char* ws = (char*)d_ws;
    size_t off = 0;
    auto carve = [&](size_t bytes) {
        void* p = ws + off;
        off = (off + bytes + 255) & ~(size_t)255;
        return p;
    };
    __bf16* xb     = (__bf16*)carve((size_t)M * DIM * 2);
    __bf16* wqkvT  = (__bf16*)carve((size_t)N3 * DIM * 2);
    __bf16* woutT  = (__bf16*)carve((size_t)DIM * DIM * 2);
    __bf16* qkvb   = (__bf16*)carve((size_t)M * N3 * 2);
    __bf16* Qb     = (__bf16*)carve((size_t)2 * NHEADS * SEQ_N * DIMH * 2);
    __bf16* Kb     = (__bf16*)carve((size_t)2 * NHEADS * SEQ_N * DIMH * 2);
    __bf16* Vt     = (__bf16*)carve((size_t)2 * NHEADS * DIMH * SEQ_N * 2);
    __bf16* attnb  = (__bf16*)carve((size_t)M * DIM * 2);

    // Stage 0: casts / weight transposes.
    cast_bf16_kernel<<<4096, 256, 0, stream>>>(x, xb, M * DIM);
    transpose_cast_kernel<<<(N3 * DIM) / 256, 256, 0, stream>>>(Wqkv, wqkvT, DIM, N3);
    transpose_cast_kernel<<<(DIM * DIM) / 256, 256, 0, stream>>>(Wout, woutT, DIM, DIM);

    // Stage 1: qkv = x @ W_qkv  (bf16 out).
    gemm_bf16_kernel<<<dim3(M / 128, N3 / 64), 256, 0, stream>>>(
        xb, wqkvT, nullptr, qkvb, M, N3, DIM);

    // Stage 2: RoPE + scatter to per-head layouts (V transposed).
    rope_scatter_kernel<<<dim3(M, NHEADS), 64, 0, stream>>>(qkvb, rot, Qb, Kb, Vt);

    // Stage 3: banded flash attention.
    attn_kernel<<<dim3(SEQ_N / 64, 2 * NHEADS), 128, 0, stream>>>(Qb, Kb, Vt, attnb);

    // Stage 4: out = attn @ W_out  (f32 out).
    gemm_bf16_kernel<<<dim3(M / 128, DIM / 64), 256, 0, stream>>>(
        attnb, woutT, out, nullptr, M, DIM, DIM);
}
